// SwinTransformerBlock_23768349016112
// MI455X (gfx1250) — compile-verified, matrix operand order checked by reading, not run on hardware
//
#include <hip/hip_runtime.h>
#include <cstdint>
#include <cstddef>

// ---------------------------------------------------------------------------
// Types
// ---------------------------------------------------------------------------
typedef __bf16 bf16;
typedef __attribute__((ext_vector_type(16))) bf16     v16bf;
typedef __attribute__((ext_vector_type(8)))  bf16     v8bf;
typedef __attribute__((ext_vector_type(8)))  float    v8f;
typedef __attribute__((ext_vector_type(4)))  unsigned uv4;
typedef __attribute__((ext_vector_type(8)))  unsigned uv8;

#define DEV static __device__ __forceinline__

// f32 -> bf16 round-to-nearest-even
DEV unsigned rtne(float f) {
    unsigned u = __builtin_bit_cast(unsigned, f);
    return u + 0x7fffu + ((u >> 16) & 1u);
}
DEV bf16 f2b(float f) {
    return __builtin_bit_cast(bf16, (unsigned short)(rtne(f) >> 16));
}
DEV unsigned pack2(float a, float b) {   // [b|a] packed bf16 pair
    return (rtne(b) & 0xffff0000u) | (rtne(a) >> 16);
}
// build a v16bf fragment from two 16-byte chunks (lo = elems 0..7, hi = 8..15)
DEV v16bf ld16(const bf16* lo, const bf16* hi) {
    v8bf a = *(const v8bf*)lo;
    v8bf b = *(const v8bf*)hi;
    return __builtin_shufflevector(a, b, 0,1,2,3,4,5,6,7,8,9,10,11,12,13,14,15);
}

// Problem constants
static constexpr int HIMG   = 56;
static constexpr int CDIM   = 384;
static constexpr int HID    = 1536;
static constexpr int NHEAD  = 12;
static constexpr int HD     = 32;
static constexpr int WIN    = 7;
static constexpr int NT     = 49;       // tokens per window
static constexpr int SHIFT  = 3;
static constexpr int NWIN   = 2048;     // 32 * 64 windows total
static constexpr int TOK    = 100352;   // 32 * 3136 tokens
static constexpr int QKVC   = 3 * CDIM; // 1152

// ---------------------------------------------------------------------------
// Kernel 1/5: LayerNorm (+ optional shift/window-partition) -> bf16
// One wave (32 lanes) per token; 12 channels per lane.
// ---------------------------------------------------------------------------
template <bool SHIFTED>
__global__ __launch_bounds__(256) void k_layernorm(const float* __restrict__ x,
                                                   const float* __restrict__ gw,
                                                   const float* __restrict__ gb,
                                                   bf16* __restrict__ out) {
    const int lane = threadIdx.x & 31;
    const int wv   = threadIdx.x >> 5;
    const int tw   = blockIdx.x * 8 + wv;     // destination token (window order if SHIFTED)

    int src;
    if (SHIFTED) {
        const int w_ = tw / NT, n = tw % NT;
        const int b_ = w_ >> 6, wi = w_ & 63;
        const int wy = wi >> 3, wx = wi & 7;
        const int iy = n / WIN, ix = n % WIN;
        const int sr = (wy * WIN + iy + SHIFT) % HIMG;
        const int sc = (wx * WIN + ix + SHIFT) % HIMG;
        src = b_ * (HIMG * HIMG) + sr * HIMG + sc;
    } else {
        src = tw;
    }

    const float4* xr = (const float4*)(x + (size_t)src * CDIM);
    float4 v0 = xr[lane * 3 + 0];
    float4 v1 = xr[lane * 3 + 1];
    float4 v2 = xr[lane * 3 + 2];

    float s = v0.x + v0.y + v0.z + v0.w + v1.x + v1.y + v1.z + v1.w
            + v2.x + v2.y + v2.z + v2.w;
    float q = v0.x*v0.x + v0.y*v0.y + v0.z*v0.z + v0.w*v0.w
            + v1.x*v1.x + v1.y*v1.y + v1.z*v1.z + v1.w*v1.w
            + v2.x*v2.x + v2.y*v2.y + v2.z*v2.z + v2.w*v2.w;
    #pragma unroll
    for (int off = 1; off < 32; off <<= 1) {
        s += __shfl_xor(s, off, 32);
        q += __shfl_xor(q, off, 32);
    }
    const float mean = s * (1.0f / CDIM);
    const float var  = q * (1.0f / CDIM) - mean * mean;
    const float rstd = rsqrtf(var + 1e-5f);

    const float4* gwr = (const float4*)gw;
    const float4* gbr = (const float4*)gb;
    uint2* op = (uint2*)(out + (size_t)tw * CDIM + lane * 12);
    #pragma unroll
    for (int j = 0; j < 3; j++) {
        float4 xv = (j == 0) ? v0 : (j == 1) ? v1 : v2;
        float4 wv4 = gwr[lane * 3 + j];
        float4 bv4 = gbr[lane * 3 + j];
        float o0 = (xv.x - mean) * rstd * wv4.x + bv4.x;
        float o1 = (xv.y - mean) * rstd * wv4.y + bv4.y;
        float o2 = (xv.z - mean) * rstd * wv4.z + bv4.z;
        float o3 = (xv.w - mean) * rstd * wv4.w + bv4.w;
        uint2 p; p.x = pack2(o0, o1); p.y = pack2(o2, o3);
        op[j] = p;
    }
}

// ---------------------------------------------------------------------------
// Generic 128x128-tile bf16 WMMA GEMM:  C = A(bf16, row-major) * B(f32 weights)
// 256 threads = 8 waves (4x2).  Each wave: 32 rows x 64 cols = 2x4 WMMA tiles.
// A tiles (128x32 bf16, strided) are DMA'd into LDS by the Tensor Data Mover
// (one tensor_load_to_lds per tile, issued by wave 0, tracked on TENSORcnt,
// double buffered against the WMMA loop).  B tiles are converted f32->bf16 and
// transposed into LDS so both fragment types are two contiguous b128 DS reads.
// ---------------------------------------------------------------------------
template <class Epi>
DEV void gemm_bf16(const bf16* __restrict__ A, int lda,
                   const float* __restrict__ Bw, int ldb,
                   int K, Epi epi) {
    __shared__ alignas(16) bf16 As[2][128 * 32];
    __shared__ alignas(16) bf16 Bs[2][128 * 32];

    const int rowBase = blockIdx.x * 128;
    const int colBase = blockIdx.y * 128;
    const int t = threadIdx.x, lane = t & 31, wid = t >> 5;
    const int wm = wid & 3, wn = wid >> 2;
    const int m16 = lane & 15, half = lane >> 4;

    v8f acc[2][4] = {};

    // --- A tile via TDM (D# built per CDNA5 ISA 8.3/8.4, 2D tile) ----------
    auto issueA = [&](int buf, int kt) {
        if (wid == 0) {
            const unsigned long long ga =
                (unsigned long long)(uintptr_t)(A + (size_t)rowBase * lda + kt * 32);
            const unsigned la = (unsigned)(uintptr_t)&As[buf][0]; // LDS byte offset
            uv4 g0;
            g0[0] = 1u;                                  // count=1, user descriptor
            g0[1] = la;                                  // lds_addr
            g0[2] = (unsigned)ga;                        // global_addr[31:0]
            g0[3] = (unsigned)(ga >> 32) | (2u << 30);   // global_addr[56:32] | type=2
            uv8 g1;
            g1[0] = 1u << 16;                            // data_size=2B, no mask/flags
            g1[1] = 32u << 16;                           // tensor_dim0 = 32
            g1[2] = 128u << 16;                          // tensor_dim1 = 128
            g1[3] = 32u << 16;                           // tile_dim0 = 32
            g1[4] = 128u;                                // tile_dim1 = 128, tile_dim2 = 0
            g1[5] = (unsigned)lda;                       // tensor_dim0_stride (elements)
            g1[6] = 0u;
            g1[7] = 0u;
            asm volatile("tensor_load_to_lds %0, %1" :: "s"(g0), "s"(g1) : "memory");
        }
    };
    // --- B tile: convert + transpose into [n][k], packed pair stores -------
    auto stageB = [&](int buf, int kt) {
        const int k0 = kt * 32;
        const int bn = t & 127, bh = t >> 7;
        const float* src = Bw + (size_t)(k0 + bh * 16) * ldb + colBase + bn;
        unsigned* dst = (unsigned*)&Bs[buf][bn * 32 + bh * 16];
        #pragma unroll
        for (int kk = 0; kk < 8; kk++) {
            float w0 = src[(size_t)(2 * kk) * ldb];
            float w1 = src[(size_t)(2 * kk + 1) * ldb];
            dst[kk] = pack2(w0, w1);
        }
    };

    const int KT = K / 32;
    int buf = 0;
    issueA(0, 0);
    stageB(0, 0);
    if (wid == 0) __builtin_amdgcn_s_wait_tensorcnt(0);
    __syncthreads();

    for (int kt = 0; kt < KT; kt++) {
        if (kt + 1 < KT) {
            issueA(buf ^ 1, kt + 1);
            stageB(buf ^ 1, kt + 1);
        }
        v16bf af[2], bfv[4];
        #pragma unroll
        for (int ms = 0; ms < 2; ms++) {
            const bf16* p = &As[buf][(wm * 32 + ms * 16 + m16) * 32 + half * 8];
            af[ms] = ld16(p, p + 16);
        }
        #pragma unroll
        for (int ns = 0; ns < 4; ns++) {
            const bf16* p = &Bs[buf][(wn * 64 + ns * 16 + m16) * 32 + half * 8];
            bfv[ns] = ld16(p, p + 16);
        }
        #pragma unroll
        for (int ms = 0; ms < 2; ms++)
            #pragma unroll
            for (int ns = 0; ns < 4; ns++)
                acc[ms][ns] = __builtin_amdgcn_wmma_f32_16x16x32_bf16(
                    false, af[ms], false, bfv[ns], (short)0, acc[ms][ns], false, false);
        if (wid == 0 && kt + 1 < KT) __builtin_amdgcn_s_wait_tensorcnt(0);
        __syncthreads();
        buf ^= 1;
    }

    #pragma unroll
    for (int ms = 0; ms < 2; ms++)
        #pragma unroll
        for (int ns = 0; ns < 4; ns++)
            #pragma unroll
            for (int r = 0; r < 8; r++) {
                const int row = rowBase + wm * 32 + ms * 16 + half * 8 + r;
                const int col = colBase + wn * 64 + ns * 16 + m16;
                epi(row, col, acc[ms][ns][r]);
            }
}

// --- GEMM wrappers -----------------------------------------------------------
// QKV: q/k thirds stored row-major; v third stored transposed per (window,head)
// as vT[w][head][d][token(pad 64)] so attention B-frags are contiguous loads.
__global__ __launch_bounds__(256) void k_gemm_qkv(const bf16* __restrict__ A,
                                                  const float* __restrict__ W,
                                                  const float* __restrict__ bias,
                                                  bf16* __restrict__ out,
                                                  bf16* __restrict__ vt) {
    gemm_bf16(A, CDIM, W, QKVC, CDIM, [&](int row, int col, float v) {
        const float z = v + bias[col];
        if (col < 2 * CDIM) {
            out[(size_t)row * QKVC + col] = f2b(z);
        } else {
            const int w_ = row / NT, n = row % NT;
            const int hc = col - 2 * CDIM;
            const int h = hc >> 5, d = hc & 31;
            vt[(((size_t)w_ * NHEAD + h) * HD + d) * 64 + n] = f2b(z);
        }
    });
}

__global__ __launch_bounds__(256) void k_gemm_proj(const bf16* __restrict__ A,
                                                   const float* __restrict__ W,
                                                   const float* __restrict__ bias,
                                                   const float* __restrict__ xin,
                                                   float* __restrict__ out) {
    gemm_bf16(A, CDIM, W, CDIM, CDIM, [&](int row, int col, float v) {
        // window order -> (batch, spatial) with inverse cyclic shift
        const int w_ = row / NT, n = row % NT;
        const int b_ = w_ >> 6, wi = w_ & 63;
        const int wy = wi >> 3, wx = wi & 7;
        const int iy = n / WIN, ix = n % WIN;
        const int fr = (wy * WIN + iy + SHIFT) % HIMG;
        const int fc = (wx * WIN + ix + SHIFT) % HIMG;
        const size_t idx = (size_t)b_ * (HIMG * HIMG) + fr * HIMG + fc;
        out[idx * CDIM + col] = xin[idx * CDIM + col] + v + bias[col];
    });
}

__global__ __launch_bounds__(256) void k_gemm_fc1(const bf16* __restrict__ A,
                                                  const float* __restrict__ W,
                                                  const float* __restrict__ bias,
                                                  bf16* __restrict__ out) {
    gemm_bf16(A, CDIM, W, HID, CDIM, [&](int row, int col, float v) {
        const float z = v + bias[col];
        const float g = 0.5f * z * (1.0f + erff(z * 0.70710678118654752f)); // exact GELU
        out[(size_t)row * HID + col] = f2b(g);
    });
}

__global__ __launch_bounds__(256) void k_gemm_fc2(const bf16* __restrict__ A,
                                                  const float* __restrict__ W,
                                                  const float* __restrict__ bias,
                                                  float* __restrict__ out) {
    gemm_bf16(A, HID, W, CDIM, HID, [&](int row, int col, float v) {
        out[(size_t)row * CDIM + col] += v + bias[col];
    });
}

// ---------------------------------------------------------------------------
// Attention: one wave per (window, head).  N=49 padded to 64.
// QK^T: 4x4 tiles, K=32 (one WMMA each).  Softmax in registers (row = 16
// lanes x 4 col-tiles).  Probs bounced through LDS to build A-frags for PV.
// V comes from the pre-transposed vT buffer -> contiguous b128 B-frag loads.
// ---------------------------------------------------------------------------
DEV int zone3(int g) { return g < 49 ? 0 : (g < 53 ? 1 : 2); }

__global__ __launch_bounds__(128) void k_attention(const bf16* __restrict__ qkv,
                                                   const bf16* __restrict__ vt,
                                                   const float* __restrict__ rpb,
                                                   bf16* __restrict__ aout) {
    __shared__ alignas(16) bf16 probs[4][64 * 64];

    const int w    = blockIdx.x;
    const int wv   = threadIdx.x >> 5;
    const int head = blockIdx.y * 4 + wv;
    const int lane = threadIdx.x & 31;
    const int m16  = lane & 15, half = lane >> 4;
    const int wi = w & 63, wy = wi >> 3, wx = wi & 7;

    const bf16* base = qkv + (size_t)w * NT * QKVC;

    // ---- Q (A-frag) and K (B-frag) loads: identical contiguous patterns ----
    v16bf qf[4], kf[4];
    #pragma unroll
    for (int tmi = 0; tmi < 4; tmi++) {
        int tok = tmi * 16 + m16; tok = tok > 48 ? 48 : tok;
        const bf16* pq = base + (size_t)tok * QKVC + head * HD;
        const bf16* pk = pq + CDIM;
        qf[tmi] = ld16(pq + half * 8, pq + 16 + half * 8);
        kf[tmi] = ld16(pk + half * 8, pk + 16 + half * 8);
    }

    // ---- scores ----
    v8f s[4][4];
    #pragma unroll
    for (int tm = 0; tm < 4; tm++)
        #pragma unroll
        for (int tn = 0; tn < 4; tn++) {
            v8f z = {};
            s[tm][tn] = __builtin_amdgcn_wmma_f32_16x16x32_bf16(
                false, qf[tm], false, kf[tn], (short)0, z, false, false);
        }

    // ---- scale + rel-pos bias + shift mask + softmax ----
    const float scale = 0.17677669529663687f; // 1/sqrt(32)
    #pragma unroll
    for (int tm = 0; tm < 4; tm++) {
        #pragma unroll
        for (int r = 0; r < 8; r++) {
            const int row = tm * 16 + half * 8 + r;
            const int y1 = row / WIN, x1 = row % WIN;
            const int regA = 3 * zone3(wy * WIN + y1) + zone3(wx * WIN + x1);
            float vals[4];
            float rmax = -3e38f;
            #pragma unroll
            for (int tn = 0; tn < 4; tn++) {
                const int col = tn * 16 + m16;
                const int y2 = col / WIN, x2 = col % WIN;
                const int regB = 3 * zone3(wy * WIN + y2) + zone3(wx * WIN + x2);
                int idx = (y1 - y2 + WIN - 1) * (2 * WIN - 1) + (x1 - x2 + WIN - 1);
                idx = idx < 0 ? 0 : (idx > 168 ? 168 : idx);
                const float bias = rpb[idx * NHEAD + head];
                const float msk  = (regA != regB) ? -100.0f : 0.0f;
                const bool valid = (row < NT) && (col < NT);
                float sv = s[tm][tn][r] * scale + bias + msk;
                sv = valid ? sv : -1e30f;
                vals[tn] = sv;
                rmax = fmaxf(rmax, sv);
            }
            #pragma unroll
            for (int off = 1; off < 16; off <<= 1)
                rmax = fmaxf(rmax, __shfl_xor(rmax, off, 32));
            float rsum = 0.0f;
            #pragma unroll
            for (int tn = 0; tn < 4; tn++) {
                vals[tn] = __expf(vals[tn] - rmax);
                rsum += vals[tn];
            }
            #pragma unroll
            for (int off = 1; off < 16; off <<= 1)
                rsum += __shfl_xor(rsum, off, 32);
            const float rinv = 1.0f / rsum;
            #pragma unroll
            for (int tn = 0; tn < 4; tn++)
                s[tm][tn][r] = vals[tn] * rinv;
        }
    }

    // ---- probs -> LDS (bf16), re-read in A-frag layout ----
    bf16* pb = probs[wv];
    #pragma unroll
    for (int tm = 0; tm < 4; tm++)
        #pragma unroll
        for (int tn = 0; tn < 4; tn++)
            #pragma unroll
            for (int r = 0; r < 8; r++)
                pb[(tm * 16 + half * 8 + r) * 64 + tn * 16 + m16] = f2b(s[tm][tn][r]);
    __syncthreads();

    // ---- V fragments from vT[w][head][d][token]: contiguous b128 loads ----
    const bf16* vtw = vt + ((size_t)(w * NHEAD + head) * HD) * 64;
    v16bf vf[2][2];
    #pragma unroll
    for (int kt = 0; kt < 2; kt++)
        #pragma unroll
        for (int ns = 0; ns < 2; ns++) {
            const bf16* p = vtw + (size_t)(ns * 16 + m16) * 64 + kt * 32;
            vf[kt][ns] = ld16(p + half * 8, p + 16 + half * 8);
        }

    // ---- PV ----
    #pragma unroll
    for (int tm = 0; tm < 4; tm++) {
        const bf16* prow = &pb[(tm * 16 + m16) * 64];
        v16bf pa0 = ld16(prow + half * 8,      prow + 16 + half * 8);
        v16bf pa1 = ld16(prow + 32 + half * 8, prow + 48 + half * 8);
        #pragma unroll
        for (int ns = 0; ns < 2; ns++) {
            v8f o = {};
            o = __builtin_amdgcn_wmma_f32_16x16x32_bf16(false, pa0, false, vf[0][ns], (short)0, o, false, false);
            o = __builtin_amdgcn_wmma_f32_16x16x32_bf16(false, pa1, false, vf[1][ns], (short)0, o, false, false);
            #pragma unroll
            for (int r = 0; r < 8; r++) {
                const int row = tm * 16 + half * 8 + r;
                if (row < NT)
                    aout[((size_t)w * NT + row) * CDIM + head * HD + ns * 16 + m16] = f2b(o[r]);
            }
        }
    }
}

// ---------------------------------------------------------------------------
// Launch
// ---------------------------------------------------------------------------
extern "C" void kernel_launch(void* const* d_in, const int* in_sizes, int n_in,
                              void* d_out, int out_size, void* d_ws, size_t ws_size,
                              hipStream_t stream) {
    const float* x       = (const float*)d_in[0];
    const float* norm1_w = (const float*)d_in[1];
    const float* norm1_b = (const float*)d_in[2];
    const float* qkv_w   = (const float*)d_in[3];
    const float* qkv_b   = (const float*)d_in[4];
    const float* rpb     = (const float*)d_in[5];
    const float* proj_w  = (const float*)d_in[6];
    const float* proj_b  = (const float*)d_in[7];
    const float* norm2_w = (const float*)d_in[8];
    const float* norm2_b = (const float*)d_in[9];
    const float* fc1_w   = (const float*)d_in[10];
    const float* fc1_b   = (const float*)d_in[11];
    const float* fc2_w   = (const float*)d_in[12];
    const float* fc2_b   = (const float*)d_in[13];
    float* out = (float*)d_out;

    // workspace layout (bytes); hidden overlays the qkv+attn slots, vt sits past
    char* ws = (char*)d_ws;
    const size_t SZ_XW  = (size_t)TOK * CDIM * sizeof(bf16);            //  77 MB
    const size_t SZ_QKV = (size_t)TOK * QKVC * sizeof(bf16);            // 231 MB
    const size_t SZ_ATT = (size_t)TOK * CDIM * sizeof(bf16);            //  77 MB
    bf16* xw     = (bf16*)(ws);                                         // LN1 windowed
    bf16* qkvbuf = (bf16*)(ws + SZ_XW);                                 // q,k (v unused)
    bf16* attno  = (bf16*)(ws + SZ_XW + SZ_QKV);                        // attn out
    bf16* vt     = (bf16*)(ws + SZ_XW + SZ_QKV + SZ_ATT);               // V transposed, 100 MB
    bf16* ln2o   = (bf16*)(ws);                                         // reuse xw slot
    bf16* hidden = (bf16*)(ws + SZ_XW);                                 // reuse qkv+attn slots

    // 1. LN1 + shift + window partition  (100352 tokens, 8 waves/block)
    k_layernorm<true><<<TOK / 8, 256, 0, stream>>>(x, norm1_w, norm1_b, xw);
    // 2. QKV GEMM: 100352x384 @ 384x1152 (v third written transposed)
    k_gemm_qkv<<<dim3(TOK / 128, QKVC / 128), 256, 0, stream>>>(xw, qkv_w, qkv_b, qkvbuf, vt);
    // 3. Windowed attention: 2048 windows x 3 head-groups (4 waves = 4 heads)
    k_attention<<<dim3(NWIN, NHEAD / 4), 128, 0, stream>>>(qkvbuf, vt, rpb, attno);
    // 4. Proj GEMM + window reverse + unshift + residual -> d_out (f32)
    k_gemm_proj<<<dim3(TOK / 128, CDIM / 128), 256, 0, stream>>>(attno, proj_w, proj_b, x, out);
    // 5. LN2 -> bf16
    k_layernorm<false><<<TOK / 8, 256, 0, stream>>>(out, norm2_w, norm2_b, ln2o);
    // 6. FC1 + GELU: 100352x384 @ 384x1536
    k_gemm_fc1<<<dim3(TOK / 128, HID / 128), 256, 0, stream>>>(ln2o, fc1_w, fc1_b, hidden);
    // 7. FC2 + bias + residual accumulate into d_out
    k_gemm_fc2<<<dim3(TOK / 128, CDIM / 128), 256, 0, stream>>>(hidden, fc2_w, fc2_b, out);
}
